// ExactSequenceAttention_68212670595410
// MI455X (gfx1250) — matrix operand order repaired and checked by reading, not compiled, not run on hardware
//
#include <hip/hip_runtime.h>
#include <hip/hip_bf16.h>
#include <cstdint>
#include <math.h>

typedef _Float16 half_t;
typedef __attribute__((ext_vector_type(16))) _Float16 v16h;
typedef __attribute__((ext_vector_type(8)))  float    v8f;

// ---------------------------------------------------------------------------
// CDNA5 async global->LDS copy (per-lane 16B), tracked by ASYNCcnt.
// GV mode: lds-byte-address VGPR, 64-bit global address VGPR pair, saddr=off.
// ---------------------------------------------------------------------------
__device__ __forceinline__ void async_copy_b128(const half_t* g, half_t* l) {
  asm volatile("global_load_async_to_lds_b128 %0, %1, off"
               :: "v"((uint32_t)(uintptr_t)l), "v"((uint64_t)(uintptr_t)g)
               : "memory");
}
__device__ __forceinline__ void wait_async0() {
  asm volatile("s_wait_asynccnt 0x0" ::: "memory");
}

// ---------------------------------------------------------------------------
// elementwise f32 -> f16
// ---------------------------------------------------------------------------
__global__ void cvt_f32_f16(const float* __restrict__ in, half_t* __restrict__ out, size_t n) {
  size_t i = (size_t)blockIdx.x * blockDim.x + threadIdx.x;
  size_t stride = (size_t)gridDim.x * blockDim.x;
  for (; i < n; i += stride) out[i] = (half_t)in[i];
}

// ---------------------------------------------------------------------------
// WMMA GEMM: C = scale * (A @ B^T) + bias
//   A: (M,K) f16 row-major, leading dim lda, per-batch stride strideA
//   B: (N,K) f16 row-major, leading dim ldb, per-batch stride strideB (0 = shared)
//   C: (M,N) OutT,          leading dim ldc, per-batch stride strideC
// grid = (N/128, M/128, batch), block = 256 threads (8 wave32)
// M, N multiples of 128; K multiple of 64 (true for all call sites here).
// ---------------------------------------------------------------------------
constexpr int BM = 128, BN = 128, BK = 64;
constexpr int KSTEPS = BK / 32;            // 2 WMMA k-steps per staged tile
constexpr int WM = 64,  WN = 32;           // per-wave tile (2x4 wave grid)
constexpr int TM = WM / 16;                // 4
constexpr int TN = WN / 16;                // 2
constexpr int LPITCH = BK + 8;             // 72 halves (144B) row pitch vs bank conflicts

template <typename OutT>
__global__ __launch_bounds__(256) void gemm_abT(
    const half_t* __restrict__ A, int64_t lda, int64_t strideA,
    const half_t* __restrict__ B, int64_t ldb, int64_t strideB,
    const float* __restrict__ bias,
    OutT* __restrict__ C, int64_t ldc, int64_t strideC,
    int K, float scale)
{
  __shared__ __align__(16) half_t As[2][BM][LPITCH];
  __shared__ __align__(16) half_t Bs[2][BN][LPITCH];

  const int tid   = threadIdx.x;
  const int wave  = tid >> 5;
  const int lane  = tid & 31;
  const int waveM = wave >> 2;      // 0..1
  const int waveN = wave & 3;       // 0..3
  const int lhalf = lane >> 4;      // 0/1 : which K-half this lane owns
  const int lmod  = lane & 15;
  const int bm0   = blockIdx.y * BM;
  const int bn0   = blockIdx.x * BN;

  A += (int64_t)blockIdx.z * strideA;
  B += (int64_t)blockIdx.z * strideB;
  C += (int64_t)blockIdx.z * strideC;

  // global -> LDS staging map: 256 threads, each moves 32 halves of A and of B
  const int ldRow = tid >> 1;            // 0..127
  const int ldK   = (tid & 1) * 32;      // 0 or 32
  const half_t* agp = A + (int64_t)(bm0 + ldRow) * lda + ldK;
  const half_t* bgp = B + (int64_t)(bn0 + ldRow) * ldb + ldK;
  half_t* asl = &As[0][ldRow][ldK];
  half_t* bsl = &Bs[0][ldRow][ldK];
  constexpr int bufStride = BM * LPITCH;   // halves per double-buffer slot

  // issue 8 async b128 copies staging one 128x64 A tile + B tile
  auto stage = [&](int kt, int buf) {
    const half_t* ag = agp + (int64_t)kt * BK;
    const half_t* bg = bgp + (int64_t)kt * BK;
    half_t* al = asl + buf * bufStride;
    half_t* bl = bsl + buf * bufStride;
#pragma unroll
    for (int j = 0; j < 4; ++j) {
      async_copy_b128(ag + j * 8, al + j * 8);
      async_copy_b128(bg + j * 8, bl + j * 8);
    }
  };

  v8f acc[TM][TN];
#pragma unroll
  for (int mi = 0; mi < TM; ++mi)
#pragma unroll
    for (int ni = 0; ni < TN; ++ni)
      acc[mi][ni] = v8f{0.f, 0.f, 0.f, 0.f, 0.f, 0.f, 0.f, 0.f};

  stage(0, 0);
  wait_async0();
  __syncthreads();

  const int ktiles = K / BK;
  for (int kt = 0; kt < ktiles; ++kt) {
    const int cur = kt & 1;
    const bool more = (kt + 1) < ktiles;
    if (more) stage(kt + 1, cur ^ 1);   // async DMA fills next buffer during compute

    union Frag { v16h v; float4 f[2]; };
#pragma unroll
    for (int ks = 0; ks < KSTEPS; ++ks) {
      const int kb = ks * 32;
      Frag afrag[TM], bfrag[TN];
#pragma unroll
      for (int mi = 0; mi < TM; ++mi) {
        const int row = waveM * WM + mi * 16 + lmod;
        // A 16x32 f16 layout: lanes 0-15 K {0..7,16..23}; lanes 16-31 K {8..15,24..31}
        afrag[mi].f[0] = *(const float4*)&As[cur][row][kb + lhalf * 8];
        afrag[mi].f[1] = *(const float4*)&As[cur][row][kb + 16 + lhalf * 8];
      }
#pragma unroll
      for (int ni = 0; ni < TN; ++ni) {
        const int col = waveN * WN + ni * 16 + lmod;
        // B 32x16 f16 layout: lanes 0-15 K {0..7,16..23}; lanes 16-31 K {8..15,24..31}
        bfrag[ni].f[0] = *(const float4*)&Bs[cur][col][kb + lhalf * 8];
        bfrag[ni].f[1] = *(const float4*)&Bs[cur][col][kb + 16 + lhalf * 8];
      }
#pragma unroll
      for (int mi = 0; mi < TM; ++mi)
#pragma unroll
        for (int ni = 0; ni < TN; ++ni)
          acc[mi][ni] = __builtin_amdgcn_wmma_f32_16x16x32_f16(
              /*neg_a=*/false, afrag[mi].v, /*neg_b=*/false, bfrag[ni].v,
              /*c_mod=*/(short)0, acc[mi][ni], /*reuse_a=*/false, /*reuse_b=*/false);
    }

    if (more) wait_async0();   // our async copies into the other buffer are done
    __syncthreads();           // publish to all waves / protect buffer reuse
  }

  // ---- epilogue: v8f layout = VGPR r -> M = r + lhalf*8, N = lmod ----
#pragma unroll
  for (int mi = 0; mi < TM; ++mi) {
    const int row0 = bm0 + waveM * WM + mi * 16 + lhalf * 8;
#pragma unroll
    for (int ni = 0; ni < TN; ++ni) {
      const int col = bn0 + waveN * WN + ni * 16 + lmod;
      const float bv = bias ? bias[col] : 0.0f;
      const v8f a = acc[mi][ni];
#pragma unroll
      for (int r = 0; r < 8; ++r) {
        const float v = a[r] * scale + bv;
        C[(int64_t)(row0 + r) * ldc + col] = (OutT)v;
      }
    }
  }
}

// ---------------------------------------------------------------------------
// Fused dual softmax + combine:
//   w[row, :] = 0.5*(softmax(s1[row,:]) + softmax(s2[row,:]*mask[:]))  -> f16
// one 256-thread block per row (row length = Ncols = 2048)
// ---------------------------------------------------------------------------
__device__ __forceinline__ float blk_reduce_max(float v, float* red) {
  const int tid = threadIdx.x;
  red[tid] = v; __syncthreads();
  for (int s = 128; s > 0; s >>= 1) {
    if (tid < s) red[tid] = fmaxf(red[tid], red[tid + s]);
    __syncthreads();
  }
  const float r = red[0]; __syncthreads();
  return r;
}
__device__ __forceinline__ float blk_reduce_sum(float v, float* red) {
  const int tid = threadIdx.x;
  red[tid] = v; __syncthreads();
  for (int s = 128; s > 0; s >>= 1) {
    if (tid < s) red[tid] += red[tid + s];
    __syncthreads();
  }
  const float r = red[0]; __syncthreads();
  return r;
}

__global__ __launch_bounds__(256) void softmax_combine(
    const float* __restrict__ s1, const float* __restrict__ s2,
    const float* __restrict__ mask, half_t* __restrict__ w, int Ncols)
{
  __shared__ float red[256];
  const int64_t row = blockIdx.x;
  const float* r1 = s1 + row * Ncols;
  const float* r2 = s2 + row * Ncols;
  half_t* wr = w + row * Ncols;
  const int tid = threadIdx.x;

  float m1 = -3.0e38f, m2 = -3.0e38f;
  for (int c = tid; c < Ncols; c += 256) {
    m1 = fmaxf(m1, r1[c]);
    m2 = fmaxf(m2, r2[c] * mask[c]);
  }
  m1 = blk_reduce_max(m1, red);
  m2 = blk_reduce_max(m2, red);

  float z1 = 0.f, z2 = 0.f;
  for (int c = tid; c < Ncols; c += 256) {
    z1 += __expf(r1[c] - m1);
    z2 += __expf(r2[c] * mask[c] - m2);
  }
  z1 = blk_reduce_sum(z1, red);
  z2 = blk_reduce_sum(z2, red);
  const float i1 = 1.0f / z1, i2 = 1.0f / z2;

  for (int c = tid; c < Ncols; c += 256) {
    const float v = 0.5f * (__expf(r1[c] - m1) * i1 + __expf(r2[c] * mask[c] - m2) * i2);
    wr[c] = (half_t)v;
  }
}

// ---------------------------------------------------------------------------
// f16 transpose: dst[c][r] = src[r][c]  (per batch)
// block (32,8), grid (cols/32, rows/32, batch)
// ---------------------------------------------------------------------------
__global__ __launch_bounds__(256) void transpose_h(
    const half_t* __restrict__ src, int64_t lds_, int64_t strideS,
    half_t* __restrict__ dst, int64_t ldd, int64_t strideD)
{
  __shared__ half_t tile[32][33];
  src += (int64_t)blockIdx.z * strideS;
  dst += (int64_t)blockIdx.z * strideD;
  const int r0 = blockIdx.y * 32, c0 = blockIdx.x * 32;
  const int tx = threadIdx.x, ty = threadIdx.y;
#pragma unroll
  for (int j = 0; j < 32; j += 8)
    tile[ty + j][tx] = src[(int64_t)(r0 + ty + j) * lds_ + (c0 + tx)];
  __syncthreads();
#pragma unroll
  for (int j = 0; j < 32; j += 8)
    dst[(int64_t)(c0 + ty + j) * ldd + (r0 + tx)] = tile[tx][ty + j];
}

// ---------------------------------------------------------------------------
// launch
// ---------------------------------------------------------------------------
extern "C" void kernel_launch(void* const* d_in, const int* in_sizes, int n_in,
                              void* d_out, int out_size, void* d_ws, size_t ws_size,
                              hipStream_t stream) {
  (void)in_sizes; (void)n_in; (void)out_size; (void)ws_size;

  const float* x        = (const float*)d_in[0];
  const float* qkv_w    = (const float*)d_in[1];
  const float* qkv_b    = (const float*)d_in[2];
  const float* proj_w   = (const float*)d_in[3];
  const float* proj_b   = (const float*)d_in[4];
  const float* sp_w     = (const float*)d_in[5];
  const float* sp_b     = (const float*)d_in[6];
  const float* kc_w     = (const float*)d_in[7];
  const float* kc_b     = (const float*)d_in[8];
  const float* ic_w     = (const float*)d_in[9];
  const float* ic_b     = (const float*)d_in[10];
  const float* seq_mask = (const float*)d_in[11];

  constexpr int Bb = 4, Nn = 2048, Cc = 2048, Ss = 2048;
  const float inv_sq = 1.0f / sqrtf(2048.0f);

  // ---- workspace carve with lifetime-based aliasing ----
  size_t cur = 0;
  auto alloc = [&](size_t bytes) -> char* {
    char* p = (char*)d_ws + cur;
    cur += (bytes + 255) & ~(size_t)255;
    return p;
  };
  half_t* xh      = (half_t*)alloc((size_t)Bb * Nn * Cc * 2);       // later: kerh
  half_t* qkv_wh  = (half_t*)alloc((size_t)3 * Cc * Cc * 2);        // later: imgh (with sp_wh)
  half_t* sp_wh   = (half_t*)alloc((size_t)Ss * Cc * 2);
  half_t* kc_wh   = (half_t*)alloc((size_t)Ss * Ss * 2);
  half_t* ic_wh   = (half_t*)alloc((size_t)Ss * Ss * 2);
  half_t* proj_wh = (half_t*)alloc((size_t)Cc * Cc * 2);
  half_t* qkvh    = (half_t*)alloc((size_t)Bb * Nn * 3 * Cc * 2);
  half_t* sqh     = (half_t*)alloc((size_t)Bb * Nn * Ss * 2);       // later: wgt
  half_t* skh     = (half_t*)alloc((size_t)Bb * Nn * Ss * 2);       // later: vT
  float*  s1      = (float*)alloc((size_t)Bb * Nn * Nn * 4);        // later: yh
  float*  s2      = (float*)alloc((size_t)Bb * Nn * Nn * 4);

  half_t* kerh = xh;                      // xh dead after qkv GEMM
  half_t* imgh = qkv_wh;                  // qkv_wh+sp_wh dead, contiguous 33.5MB
  half_t* wgt  = sqh;                     // sqh dead after ker GEMM
  half_t* vT   = skh;                     // skh dead after img GEMM
  half_t* yh   = (half_t*)s1;             // s1 dead after softmax

  // ---- f32 -> f16 conversions ----
  cvt_f32_f16<<<4096, 256, 0, stream>>>(x,      xh,      (size_t)Bb * Nn * Cc);
  cvt_f32_f16<<<4096, 256, 0, stream>>>(qkv_w,  qkv_wh,  (size_t)3 * Cc * Cc);
  cvt_f32_f16<<<2048, 256, 0, stream>>>(sp_w,   sp_wh,   (size_t)Ss * Cc);
  cvt_f32_f16<<<2048, 256, 0, stream>>>(kc_w,   kc_wh,   (size_t)Ss * Ss);
  cvt_f32_f16<<<2048, 256, 0, stream>>>(ic_w,   ic_wh,   (size_t)Ss * Ss);
  cvt_f32_f16<<<2048, 256, 0, stream>>>(proj_w, proj_wh, (size_t)Cc * Cc);

  const dim3 blk(256);
  // ---- qkv = x @ qkv_w^T + qkv_b -> (B, N, 3C) f16 ----
  gemm_abT<half_t><<<dim3(3 * Cc / BN, Nn / BM, Bb), blk, 0, stream>>>(
      xh, Cc, (int64_t)Nn * Cc, qkv_wh, Cc, 0, qkv_b,
      qkvh, 3 * Cc, (int64_t)Nn * 3 * Cc, Cc, 1.0f);

  half_t* q = qkvh;            // column slices of qkv, row stride 3C
  half_t* k = qkvh + Cc;
  half_t* v = qkvh + 2 * Cc;

  // ---- seq_q = q @ sp_w^T + sp_b ; seq_k = k @ sp_w^T + sp_b ----
  gemm_abT<half_t><<<dim3(Ss / BN, Nn / BM, Bb), blk, 0, stream>>>(
      q, 3 * Cc, (int64_t)Nn * 3 * Cc, sp_wh, Cc, 0, sp_b,
      sqh, Ss, (int64_t)Nn * Ss, Cc, 1.0f);
  gemm_abT<half_t><<<dim3(Ss / BN, Nn / BM, Bb), blk, 0, stream>>>(
      k, 3 * Cc, (int64_t)Nn * 3 * Cc, sp_wh, Cc, 0, sp_b,
      skh, Ss, (int64_t)Nn * Ss, Cc, 1.0f);

  // ---- ker = seq_q @ kc_w^T + kc_b ; img = seq_k @ ic_w^T + ic_b ----
  gemm_abT<half_t><<<dim3(Ss / BN, Nn / BM, Bb), blk, 0, stream>>>(
      sqh, Ss, (int64_t)Nn * Ss, kc_wh, Ss, 0, kc_b,
      kerh, Ss, (int64_t)Nn * Ss, Ss, 1.0f);
  gemm_abT<half_t><<<dim3(Ss / BN, Nn / BM, Bb), blk, 0, stream>>>(
      skh, Ss, (int64_t)Nn * Ss, ic_wh, Ss, 0, ic_b,
      imgh, Ss, (int64_t)Nn * Ss, Ss, 1.0f);

  // ---- s1 = q @ k^T / sqrt(hd) ; s2 = ker @ img^T / sqrt(S)  (f32) ----
  gemm_abT<float><<<dim3(Nn / BN, Nn / BM, Bb), blk, 0, stream>>>(
      q, 3 * Cc, (int64_t)Nn * 3 * Cc, k, 3 * Cc, (int64_t)Nn * 3 * Cc, nullptr,
      s1, Nn, (int64_t)Nn * Nn, Cc, inv_sq);
  gemm_abT<float><<<dim3(Nn / BN, Nn / BM, Bb), blk, 0, stream>>>(
      kerh, Ss, (int64_t)Nn * Ss, imgh, Ss, (int64_t)Nn * Ss, nullptr,
      s2, Nn, (int64_t)Nn * Nn, Ss, inv_sq);

  // ---- final_weights = 0.5*(softmax(s1) + softmax(s2*mask)) -> f16 ----
  softmax_combine<<<dim3(Bb * Nn), blk, 0, stream>>>(s1, s2, seq_mask, wgt, Nn);

  // ---- vT[d][m] = v[m][d] per batch ----
  transpose_h<<<dim3(Cc / 32, Nn / 32, Bb), dim3(32, 8), 0, stream>>>(
      v, 3 * Cc, (int64_t)Nn * 3 * Cc, vT, Nn, (int64_t)Cc * Nn);

  // ---- y = weights @ v  == wgt (N,N) @ vT(C,N)^T -> (B,N,C) f16 ----
  gemm_abT<half_t><<<dim3(Cc / BN, Nn / BM, Bb), blk, 0, stream>>>(
      wgt, Nn, (int64_t)Nn * Nn, vT, Nn, (int64_t)Cc * Nn, nullptr,
      yh, Cc, (int64_t)Nn * Cc, Nn, 1.0f);

  // ---- out = y @ proj_w^T + proj_b -> (B,N,C) f32 ----
  gemm_abT<float><<<dim3(Cc / BN, Nn / BM, Bb), blk, 0, stream>>>(
      yh, Cc, (int64_t)Nn * Cc, proj_wh, Cc, 0, proj_b,
      (float*)d_out, Cc, (int64_t)Nn * Cc, Cc, 1.0f);
}
